// GATLayer_33749853012162
// MI455X (gfx1250) — matrix operand order repaired and checked by reading, not compiled, not run on hardware
//
#include <hip/hip_runtime.h>
#include <hip/hip_bf16.h>
#include <float.h>

typedef __attribute__((ext_vector_type(16))) _Float16 v16h;
typedef __attribute__((ext_vector_type(8)))  _Float16 v8h;
typedef __attribute__((ext_vector_type(8)))  float    v8f;

#define GAT_IN   128
#define GAT_HC   64
#define GAT_H    4
#define GAT_C    16
#define LEAKY    0.2f
#define GAT_EPS  1e-16f

// LDS row pitch for the x tile (in halves); 136*2 = 272 B (16B-aligned rows)
#define XS  (GAT_IN + 8)
// LDS row pitch for the staged f32 result tile (in floats)
#define HS  (GAT_HC + 4)

// ---------------------------------------------------------------------------
// Init: out = bias (scatter-add target), maxper = -FLT_MAX, denom = 0
// ---------------------------------------------------------------------------
__global__ __launch_bounds__(256) void gat_init(float* __restrict__ out,
                                                const float* __restrict__ bias,
                                                float* __restrict__ maxper,
                                                float* __restrict__ denom, int N)
{
    int i = blockIdx.x * 256 + threadIdx.x;
    if (i < N * GAT_HC) out[i] = bias[i & (GAT_HC - 1)];
    if (i < N * GAT_H) { maxper[i] = -FLT_MAX; denom[i] = 0.0f; }
}

// ---------------------------------------------------------------------------
// WMMA GEMM: h = x @ W  (f16 multiplicands, f32 accumulate)
// Block = 256 threads = 8 wave32 waves; block tile = 32(M) x 64(N).
// W lives in LDS pre-swizzled fragment-major (Wfrag[kt][nt][lane][0:15]) so
// each B fragment is two ds_load_b128; A fragments are two contiguous 16B
// chunks of the row-major x tile (also ds_load_b128).
// Epilogue: stage the 32x64 f32 tile in LDS, then (a) coalesced float4
// global writeback of h, (b) 128 threads compute alpha_src/alpha_dst as
// plain 16-wide dot products from LDS (no cross-lane shuffles).
// ---------------------------------------------------------------------------
__global__ __launch_bounds__(256) void gat_gemm(const float* __restrict__ x,
                                                const float* __restrict__ W,
                                                const float* __restrict__ a_src,
                                                const float* __restrict__ a_dst,
                                                float* __restrict__ h,
                                                float* __restrict__ alpha_src,
                                                float* __restrict__ alpha_dst,
                                                int N)
{
    // 16 fragments (4 k-tiles x 4 n-tiles) x 32 lanes x 16 halves = 16 KB
    __shared__ alignas(32) _Float16 Wfrag[16 * 32 * 16];
    __shared__ alignas(16) _Float16 Xl[32 * XS];     // 8.7 KB
    __shared__ alignas(16) float    Ht[32 * HS];     // 8.7 KB staged result

    const int tid  = threadIdx.x;
    const int row0 = blockIdx.x * 32;

    // Fill Wfrag (swizzle + f32->f16). 8192 elems / 256 threads = 32 each.
    for (int idx = tid; idx < 16 * 32 * 16; idx += 256) {
        int i    = idx & 15;          // element within fragment
        int ln   = (idx >> 4) & 31;   // lane
        int frag = idx >> 9;          // kt*4 + nt
        int nt   = frag & 3;
        int kt   = frag >> 2;
        int K    = kt * 32 + (ln >> 4) * 16 + i;
        int col  = nt * GAT_C + (ln & 15);
        Wfrag[idx] = (_Float16)W[K * GAT_HC + col];
    }
    // Load + convert 32 rows of x to LDS (zero-pad past N)
    for (int i = tid; i < 32 * GAT_IN; i += 256) {
        int r = i >> 7, c = i & 127;
        int row = row0 + r;
        Xl[r * XS + c] = (row < N) ? (_Float16)x[(size_t)row * GAT_IN + c]
                                   : (_Float16)0.0f;
    }
    __syncthreads();

    const int wid  = tid >> 5;
    const int lane = tid & 31;
    const int hf   = lane >> 4;   // lane half (0/1)
    const int l16  = lane & 15;
    const int mt   = wid >> 2;    // M tile: 0..1
    const int nt   = wid & 3;     // N tile == head: 0..3

    const _Float16* arow = &Xl[(mt * 16 + l16) * XS];

    v8f acc = {};
    #pragma unroll
    for (int kt = 0; kt < 4; ++kt) {
        const int k0 = kt * 32;
        // A fragment: row l16; K=k0+16*hf+i
        v8h a0 = *(const v8h*)&arow[k0 + 16 * hf];
        v8h a1 = *(const v8h*)&arow[k0 + 16 * hf + 8];
        v16h a = __builtin_shufflevector(a0, a1, 0, 1, 2, 3, 4, 5, 6, 7,
                                                 8, 9, 10, 11, 12, 13, 14, 15);
        // B fragment: one contiguous 32-byte run per lane
        v16h b = *(const v16h*)&Wfrag[((kt * 4 + nt) * 32 + lane) * 16];

        acc = __builtin_amdgcn_wmma_f32_16x16x32_f16(
                  false, a, false, b, (short)0, acc, false, false);
    }

    // Stage result tile: D layout has VGPR r at (M = r + 8*hf, N = l16)
    #pragma unroll
    for (int r = 0; r < 8; ++r) {
        int rl = mt * 16 + 8 * hf + r;
        Ht[rl * HS + nt * GAT_C + l16] = acc[r];
    }
    __syncthreads();

    // (a) Coalesced h writeback: 32 rows x 16 float4 = 512 vectors
    for (int i = tid; i < 32 * (GAT_HC / 4); i += 256) {
        int rl  = i >> 4;          // row in tile
        int c4  = i & 15;          // float4 column
        int row = row0 + rl;
        if (row < N) {
            const float* p = &Ht[rl * HS + c4 * 4];
            float4 v = make_float4(p[0], p[1], p[2], p[3]);
            ((float4*)h)[(size_t)row * (GAT_HC / 4) + c4] = v;
        }
    }

    // (b) alpha_src / alpha_dst: one (row, head) pair per thread
    if (tid < 32 * GAT_H) {
        int rl  = tid >> 2;        // row in tile
        int hd  = tid & 3;         // head
        int row = row0 + rl;
        if (row < N) {
            const float* hp = &Ht[rl * HS + hd * GAT_C];
            const float* as = &a_src[hd * GAT_C];
            const float* ad = &a_dst[hd * GAT_C];
            float s = 0.0f, d = 0.0f;
            #pragma unroll
            for (int c = 0; c < GAT_C; ++c) {
                s += hp[c] * as[c];
                d += hp[c] * ad[c];
            }
            alpha_src[row * GAT_H + hd] = s;
            alpha_dst[row * GAT_H + hd] = d;
        }
    }
}

// ---------------------------------------------------------------------------
// Float atomic max via signed/unsigned integer punning
// ---------------------------------------------------------------------------
__device__ __forceinline__ void atomicMaxF32(float* addr, float val)
{
    if (val >= 0.0f) atomicMax((int*)addr, __float_as_int(val));
    else             atomicMin((unsigned int*)addr, __float_as_uint(val));
}

__device__ __forceinline__ float leaky(float a)
{
    return (a >= 0.0f) ? a : LEAKY * a;
}

// ---------------------------------------------------------------------------
// Edge pass 1: segment max of leaky-relu logits (per dst, per head)
// ---------------------------------------------------------------------------
__global__ __launch_bounds__(256) void gat_edge_max(const int* __restrict__ ei,
                                                    const float* __restrict__ asrc,
                                                    const float* __restrict__ adst,
                                                    float* __restrict__ maxper, int E)
{
    int e = blockIdx.x * 256 + threadIdx.x;
    if (e >= E) return;
    int s = ei[e];
    int d = ei[E + e];
    #pragma unroll
    for (int hh = 0; hh < GAT_H; ++hh) {
        float a = leaky(asrc[s * GAT_H + hh] + adst[d * GAT_H + hh]);
        atomicMaxF32(&maxper[d * GAT_H + hh], a);
    }
}

// ---------------------------------------------------------------------------
// Edge pass 2: segment sum of exp(alpha - max)
// ---------------------------------------------------------------------------
__global__ __launch_bounds__(256) void gat_edge_sum(const int* __restrict__ ei,
                                                    const float* __restrict__ asrc,
                                                    const float* __restrict__ adst,
                                                    const float* __restrict__ maxper,
                                                    float* __restrict__ denom, int E)
{
    int e = blockIdx.x * 256 + threadIdx.x;
    if (e >= E) return;
    int s = ei[e];
    int d = ei[E + e];
    #pragma unroll
    for (int hh = 0; hh < GAT_H; ++hh) {
        float a  = leaky(asrc[s * GAT_H + hh] + adst[d * GAT_H + hh]);
        float ex = __expf(a - maxper[d * GAT_H + hh]);
        atomicAdd(&denom[d * GAT_H + hh], ex);
    }
}

// ---------------------------------------------------------------------------
// Edge pass 3: out[dst] += attn * h[src]   (L2-resident scatter-add)
// ---------------------------------------------------------------------------
__global__ __launch_bounds__(256) void gat_edge_agg(const int* __restrict__ ei,
                                                    const float* __restrict__ asrc,
                                                    const float* __restrict__ adst,
                                                    const float* __restrict__ maxper,
                                                    const float* __restrict__ denom,
                                                    const float* __restrict__ hbuf,
                                                    float* __restrict__ out, int E)
{
    int e = blockIdx.x * 256 + threadIdx.x;
    if (e >= E) return;
    int s = ei[e];
    int d = ei[E + e];
    const float4* hs = (const float4*)(hbuf + (size_t)s * GAT_HC);
    float* od = out + (size_t)d * GAT_HC;
    #pragma unroll
    for (int hh = 0; hh < GAT_H; ++hh) {
        float a    = leaky(asrc[s * GAT_H + hh] + adst[d * GAT_H + hh]);
        float attn = __expf(a - maxper[d * GAT_H + hh]) /
                     fmaxf(denom[d * GAT_H + hh], GAT_EPS);
        #pragma unroll
        for (int j = 0; j < 4; ++j) {
            float4 v = hs[hh * 4 + j];
            atomicAdd(od + hh * GAT_C + j * 4 + 0, v.x * attn);
            atomicAdd(od + hh * GAT_C + j * 4 + 1, v.y * attn);
            atomicAdd(od + hh * GAT_C + j * 4 + 2, v.z * attn);
            atomicAdd(od + hh * GAT_C + j * 4 + 3, v.w * attn);
        }
    }
}

// ---------------------------------------------------------------------------
extern "C" void kernel_launch(void* const* d_in, const int* in_sizes, int n_in,
                              void* d_out, int out_size, void* d_ws, size_t ws_size,
                              hipStream_t stream)
{
    const float* x    = (const float*)d_in[0];
    const int*   ei   = (const int*)d_in[1];
    const float* W    = (const float*)d_in[2];
    const float* a_s  = (const float*)d_in[3];
    const float* a_d  = (const float*)d_in[4];
    const float* bias = (const float*)d_in[5];

    const int N = in_sizes[0] / GAT_IN;
    const int E = in_sizes[1] / 2;
    float* out = (float*)d_out;

    // workspace layout (floats): h[N*64] | asrc[N*4] | adst[N*4] | max[N*4] | denom[N*4]
    float* hbuf   = (float*)d_ws;
    float* asrc   = hbuf   + (size_t)N * GAT_HC;
    float* adst   = asrc   + (size_t)N * GAT_H;
    float* maxper = adst   + (size_t)N * GAT_H;
    float* denom  = maxper + (size_t)N * GAT_H;

    gat_init<<<(N * GAT_HC + 255) / 256, 256, 0, stream>>>(out, bias, maxper, denom, N);
    gat_gemm<<<(N + 31) / 32, 256, 0, stream>>>(x, W, a_s, a_d, hbuf, asrc, adst, N);

    const int eb = (E + 255) / 256;
    gat_edge_max<<<eb, 256, 0, stream>>>(ei, asrc, adst, maxper, E);
    gat_edge_sum<<<eb, 256, 0, stream>>>(ei, asrc, adst, maxper, denom, E);
    gat_edge_agg<<<eb, 256, 0, stream>>>(ei, asrc, adst, maxper, denom, hbuf, out, E);
}
